// ETAttentionCore_25039659336015
// MI455X (gfx1250) — matrix operand order repaired and checked by reading, not compiled, not run on hardware
//
#include <hip/hip_runtime.h>
#include <hip/hip_bf16.h>
#include <stdint.h>

// Problem constants
#define BB 4
#define NN 1024
#define DD 512
#define HH 8
#define ZZ 64

typedef __attribute__((ext_vector_type(8)))  __bf16 v8bf;
typedef __attribute__((ext_vector_type(16))) __bf16 v16bf;
typedef __attribute__((ext_vector_type(2)))  __bf16 v2bf;
typedef __attribute__((ext_vector_type(8)))  float  v8f;

// ---- workspace layout (bytes) ----
// gbf : [4096, 512] bf16            = 4 MiB
// wbf : [1024, 512] bf16 (Wq;Wk)    = 1 MiB
// qbf : [B,H,N,Z] bf16              = 4 MiB
// kbf : [B,H,N,Z] bf16              = 4 MiB
// lse : [B,N,8] f32                 = 128 KiB
#define GBF_OFF  ((size_t)0)
#define WBF_OFF  (GBF_OFF + (size_t)4096*512*2)
#define QBF_OFF  (WBF_OFF + (size_t)1024*512*2)
#define KBF_OFF  (QBF_OFF + (size_t)BB*HH*NN*ZZ*2)
#define LSE_OFF  (KBF_OFF + (size_t)BB*HH*NN*ZZ*2)

static __device__ __forceinline__ v16bf cat16(v8bf a, v8bf b) {
  return __builtin_shufflevector(a, b, 0,1,2,3,4,5,6,7,8,9,10,11,12,13,14,15);
}

// -------- K0: convert inputs to bf16, zero LSE accumulator --------
__global__ __launch_bounds__(256) void k0_prep(const float* __restrict__ g,
                                               const float* __restrict__ Wq,
                                               const float* __restrict__ Wk,
                                               char* __restrict__ ws) {
  __bf16* gbf = (__bf16*)(ws + GBF_OFF);
  __bf16* wbf = (__bf16*)(ws + WBF_OFF);
  float*  lse = (float*)(ws + LSE_OFF);
  const unsigned i = blockIdx.x * blockDim.x + threadIdx.x;
  const unsigned GN = 4096u * 512u;   // g elements
  const unsigned WN = 512u * 512u;    // Wq (= Wk) elements
  if (i < GN) gbf[i] = (__bf16)g[i];
  if (i < WN) { wbf[i] = (__bf16)Wq[i]; wbf[WN + i] = (__bf16)Wk[i]; }
  if (i < (unsigned)(BB * NN * HH)) lse[i] = 0.f;
}

// -------- K1: projection GEMM  out[f, c] = sum_d W[f,d] * g[c,d] --------
// M = f (feature, 0..1023: q then k), N = c (flattened b*n, 0..4095), K = d.
// One wave per 16x16 output tile; 16 K-steps of wmma_f32_16x16x32_bf16.
__global__ __launch_bounds__(256) void k1_project(char* __restrict__ ws) {
  const __bf16* gbf = (const __bf16*)(ws + GBF_OFF);
  const __bf16* wbf = (const __bf16*)(ws + WBF_OFF);
  __bf16* qbf = (__bf16*)(ws + QBF_OFF);
  __bf16* kbf = (__bf16*)(ws + KBF_OFF);

  const unsigned wave = (blockIdx.x * blockDim.x + threadIdx.x) >> 5;
  const unsigned lane = threadIdx.x & 31u;
  const unsigned lo = lane & 15u, hi = lane >> 4;

  const unsigned ft = wave >> 8;        // 0..63
  const unsigned ct = wave & 255u;      // 0..255
  const unsigned f0 = ft * 16u;
  const unsigned c0 = ct * 16u;

  const unsigned frow = f0 + lo;        // A-matrix row (lane 0-15 = M)
  const unsigned ccol = c0 + lo;        // B-matrix col (lane%16 = N)

  v8f acc = {};
  #pragma unroll
  for (int kk = 0; kk < 16; ++kk) {
    const unsigned d0 = (unsigned)kk * 32u;
    // A: lane holds K = {0..7,16..23} (+8 if hi) of row frow
    const __bf16* ap = wbf + (size_t)frow * DD + d0 + 8u * hi;
    v16bf A = cat16(*(const v8bf*)ap, *(const v8bf*)(ap + 16));
    // B: lane holds K = {0..7,16..23} (+8 if hi) of column ccol
    const __bf16* bp = gbf + (size_t)ccol * DD + d0 + 8u * hi;
    v16bf Bv = cat16(*(const v8bf*)bp, *(const v8bf*)(bp + 16));
    acc = __builtin_amdgcn_wmma_f32_16x16x32_bf16(false, A, false, Bv,
                                                  (short)0, acc, false, false);
  }

  // D layout: lane -> (N = ccol, M = f0 + v + 8*hi). Store bf16 to [B,H,N,Z].
  const unsigned bcol = ccol >> 10;
  const unsigned ncol = ccol & 1023u;
  const bool isQ = (f0 < 512u);
  __bf16* dst = isQ ? qbf : kbf;
  const unsigned fb = isQ ? f0 : (f0 - 512u);
  #pragma unroll
  for (int v = 0; v < 8; v += 2) {
    const unsigned f = fb + (unsigned)v + 8u * hi;
    const unsigned h = f >> 6, z = f & 63u;        // z even -> 4B aligned pair
    v2bf p; p[0] = (__bf16)acc[v]; p[1] = (__bf16)acc[v + 1];
    *(v2bf*)(dst + (((size_t)bcol * HH + h) * NN + ncol) * ZZ + z) = p;
  }
}

// -------- K2: logits + head-mix + modulation + masked exp-sum --------
// One wave per (b, n-tile, m-chunk of 8 m-tiles). S = Q Kt per head via 2 WMMAs.
__global__ __launch_bounds__(256) void k2_attn(const float* __restrict__ Hw,
                                               const float* __restrict__ betas,
                                               const float* __restrict__ mod,
                                               const int* __restrict__ sizes,
                                               char* __restrict__ ws) {
  const __bf16* qbf = (const __bf16*)(ws + QBF_OFF);
  const __bf16* kbf = (const __bf16*)(ws + KBF_OFF);
  float* lse = (float*)(ws + LSE_OFF);

  const unsigned wave = (blockIdx.x * blockDim.x + threadIdx.x) >> 5;
  const unsigned lane = threadIdx.x & 31u;
  const unsigned lo = lane & 15u, hi = lane >> 4;

  const unsigned b  = wave >> 9;          // / (64*8)
  const unsigned nt = (wave >> 3) & 63u;
  const unsigned mc = wave & 7u;
  const unsigned n0 = nt * 16u;
  const int size = sizes[b];
  if ((int)n0 >= size) return;            // whole n-tile masked -> lse rows stay 0

  // mixw[h][g] = beta[h] * Hw[h][g]  (uniform -> scalar loads)
  float mixw[8][8];
  #pragma unroll
  for (int h = 0; h < 8; ++h) {
    const float bh = betas[h];
    #pragma unroll
    for (int g = 0; g < 8; ++g) mixw[h][g] = bh * Hw[h * 8 + g];
  }

  float psum[8][8];
  #pragma unroll
  for (int v = 0; v < 8; ++v)
    #pragma unroll
    for (int g = 0; g < 8; ++g) psum[v][g] = 0.f;

  #pragma unroll 1
  for (unsigned mt = mc * 8u; mt < mc * 8u + 8u; ++mt) {
    const unsigned m0 = mt * 16u;
    if ((int)m0 >= size) break;           // all later m-tiles masked too
    const unsigned m = m0 + lo;

    v8f S[8];
    #pragma unroll
    for (int h = 0; h < 8; ++h) {
      const __bf16* qb = qbf + (((size_t)b * HH + h) * NN + (n0 + lo)) * ZZ;
      const __bf16* kb = kbf + (((size_t)b * HH + h) * NN + m) * ZZ;
      v8f c = {};
      #pragma unroll
      for (int zh = 0; zh < 64; zh += 32) {
        const __bf16* ap = qb + zh + 8u * hi;
        v16bf A = cat16(*(const v8bf*)ap, *(const v8bf*)(ap + 16));
        const __bf16* bp = kb + zh + 8u * hi;
        v16bf Bv = cat16(*(const v8bf*)bp, *(const v8bf*)(bp + 16));
        c = __builtin_amdgcn_wmma_f32_16x16x32_bf16(false, A, false, Bv,
                                                    (short)0, c, false, false);
      }
      S[h] = c;
    }

    // D layout: lane -> (m = m0+lo, n = n0 + v + 8*hi)
    #pragma unroll
    for (int v = 0; v < 8; ++v) {
      const unsigned n = n0 + (unsigned)v + 8u * hi;
      const float* mp = mod + (((size_t)b * NN + n) * NN + m) * HH;
      const float4 mA = *(const float4*)mp;
      const float4 mB = *(const float4*)(mp + 4);
      const float mv[8] = {mA.x, mA.y, mA.z, mA.w, mB.x, mB.y, mB.z, mB.w};
      const bool valid = ((int)n < size) & ((int)m < size);
      #pragma unroll
      for (int g = 0; g < 8; ++g) {
        float x = 0.f;
        #pragma unroll
        for (int h = 0; h < 8; ++h) x = fmaf(mixw[h][g], S[h][v], x);
        x *= mv[g];
        psum[v][g] += valid ? __expf(x) : 0.f;   // masked -> contributes 0
      }
    }
  }

  // reduce over the 16 m-lanes, then one lane per g does the atomic add
  #pragma unroll
  for (int v = 0; v < 8; ++v) {
    const unsigned n = n0 + (unsigned)v + 8u * hi;
    #pragma unroll
    for (int g = 0; g < 8; ++g) {
      float s = psum[v][g];
      s += __shfl_xor(s, 1, 16);
      s += __shfl_xor(s, 2, 16);
      s += __shfl_xor(s, 4, 16);
      s += __shfl_xor(s, 8, 16);
      if (lo == (unsigned)g)
        atomicAdd(lse + ((size_t)b * NN + n) * HH + g, s);
    }
  }
}

// -------- K3: e = -sum_{b, n<size_b, g} log(sum_exp)/beta_g --------
__global__ __launch_bounds__(256) void k3_final(const float* __restrict__ betas,
                                                const int* __restrict__ sizes,
                                                const char* __restrict__ ws,
                                                float* __restrict__ out) {
  const float* lse = (const float*)(ws + LSE_OFF);
  __shared__ float red[256];
  float s = 0.f;
  for (unsigned i = threadIdx.x; i < (unsigned)(BB * NN * HH); i += 256u) {
    const unsigned b = i >> 13;            // / (1024*8)
    const unsigned n = (i >> 3) & 1023u;
    const unsigned g = i & 7u;
    if ((int)n < sizes[b]) s += __logf(lse[i]) / betas[g];
  }
  red[threadIdx.x] = s;
  __syncthreads();
  for (unsigned k = 128; k > 0; k >>= 1) {
    if (threadIdx.x < k) red[threadIdx.x] += red[threadIdx.x + k];
    __syncthreads();
  }
  if (threadIdx.x == 0) out[0] = -red[0];
}

extern "C" void kernel_launch(void* const* d_in, const int* in_sizes, int n_in,
                              void* d_out, int out_size, void* d_ws, size_t ws_size,
                              hipStream_t stream) {
  const float* g   = (const float*)d_in[0];
  const float* Wq  = (const float*)d_in[1];
  const float* Wk  = (const float*)d_in[2];
  const float* Hw  = (const float*)d_in[3];
  const float* be  = (const float*)d_in[4];
  const float* mod = (const float*)d_in[5];
  const int*   sz  = (const int*)d_in[6];
  char* ws = (char*)d_ws;
  float* out = (float*)d_out;

  // K0: 8192*256 = 2,097,152 threads covers the largest conversion
  k0_prep<<<8192, 256, 0, stream>>>(g, Wq, Wk, ws);
  // K1: 64 f-tiles * 256 col-tiles = 16384 waves = 2048 blocks of 8 waves
  k1_project<<<2048, 256, 0, stream>>>(ws);
  // K2: 4 * 64 * 8 = 2048 waves = 256 blocks of 8 waves
  k2_attn<<<256, 256, 0, stream>>>(Hw, be, mod, sz, ws);
  // K3: single-block final reduction writes the scalar
  k3_final<<<1, 256, 0, stream>>>(be, sz, ws, out);
}
